// ProtoConcept_14654428414810
// MI455X (gfx1250) — matrix operand order repaired and checked by reading, not compiled, not run on hardware
//
#include <hip/hip_runtime.h>
#include <math.h>

// ---------------- types ----------------
typedef __attribute__((ext_vector_type(16))) __bf16        bf16x16;
typedef __attribute__((ext_vector_type(8)))  float         f32x8;
typedef __attribute__((ext_vector_type(4)))  float         f32x4;
typedef __attribute__((ext_vector_type(4)))  unsigned int  u32x4;

// ---------------- problem constants ----------------
#define D_    64
#define H_    128
#define E_    8
#define A_    18
#define JTOT  144         // E_*A_ = 9*16 exactly
#define NTJ   9           // JTOT/16

#define NW        4       // waves per block
#define NTHREADS  (NW*32)

// ws layout (bytes)
#define OFF_W1   0        // 8192  ushort (bf16)  W1 swizzled  [8 ntile][2 ks][2 kh][16 n][16]
#define OFF_W2   16384    // 2048  ushort         W2 swizzled  [4 ks][2 kh][16 n][16] (cols 8..15 zero)
#define OFF_WE   20480    // 9216  ushort         We swizzled  [9 ntile][2 ks][2 kh][16 n][16]
#define OFF_SC   38912    // 128   float          BN scale
#define OFF_BI   39424    // 128   float          BN bias (b1 folded)
#define OFF_B2   39936    // 16    float          b2 padded
#define WBYTES   40000

// ---------------- helpers ----------------
__device__ __forceinline__ unsigned short f2bf(float f) {
  unsigned int u = __float_as_uint(f);
  u += 0x7fffu + ((u >> 16) & 1u);          // round-to-nearest-even
  return (unsigned short)(u >> 16);
}

// B fragment offset (in ushorts): generic layout ((t2*2 + kh)*16 + n)*16
__device__ __forceinline__ int boff(int t2, int kh, int n) {
  return ((t2 * 2 + kh) * 16 + n) * 16;
}

union FragU {
  unsigned short us[16];
  u32x4          q[2];
  bf16x16        v;
};

// load 16 contiguous bf16 (32B) from LDS as a B/A fragment
__device__ __forceinline__ bf16x16 load_frag16(const unsigned short* p) {
  FragU u;
  u.q[0] = ((const u32x4*)p)[0];
  u.q[1] = ((const u32x4*)p)[1];
  return u.v;
}

// Build 16-bit A-fragment from one fp32 row of X.
// element i: K = kbase + (i>=8 ? 16 : 0) + khalf*8 + (i&7)
__device__ __forceinline__ bf16x16 make_afrag(const float* xr, int kbase, int khalf) {
  const f32x4* p0 = (const f32x4*)(xr + kbase + khalf * 8);
  const f32x4* p1 = (const f32x4*)(xr + kbase + 16 + khalf * 8);
  f32x4 f0 = p0[0], f1 = p0[1];
  f32x4 f2 = p1[0], f3 = p1[1];
  FragU u;
#pragma unroll
  for (int k = 0; k < 4; ++k) {
    u.us[k]      = f2bf(f0[k]);
    u.us[4 + k]  = f2bf(f1[k]);
    u.us[8 + k]  = f2bf(f2[k]);
    u.us[12 + k] = f2bf(f3[k]);
  }
  return u.v;
}

// A-fragment of h from LDS (bf16, row-major [16][128])
__device__ __forceinline__ bf16x16 load_afrag_lds(const unsigned short* sH, int m, int ks, int kh) {
  const unsigned short* base = sH + m * H_;
  int k0 = ks * 32 + kh * 8;
  FragU u;
  u.q[0] = ((const u32x4*)(base + k0))[0];       // elements 0..7  (K = k0..k0+7)
  u.q[1] = ((const u32x4*)(base + k0 + 16))[0];  // elements 8..15 (K = k0+16..)
  return u.v;
}

// async 16B copy global -> LDS, tracked by ASYNCcnt (no VGPR data path)
__device__ __forceinline__ void async_copy16(unsigned lds_off, const void* gptr) {
  asm volatile("global_load_async_to_lds_b128 %0, %1, off"
               :
               : "v"(lds_off), "v"((unsigned long long)gptr)
               : "memory");
}

__device__ __forceinline__ void wait_async0() {
  asm volatile("s_wait_asynccnt 0" ::: "memory");
}

// ---------------- prep kernel: swizzle weights to bf16 B-fragment layout ----------------
__global__ void moe_prep(const float* __restrict__ W1, const float* __restrict__ b1,
                         const float* __restrict__ gamma, const float* __restrict__ beta,
                         const float* __restrict__ rmean, const float* __restrict__ rvar,
                         const float* __restrict__ W2, const float* __restrict__ b2,
                         const float* __restrict__ We, unsigned char* __restrict__ ws) {
  int idx = blockIdx.x * blockDim.x + threadIdx.x;
  unsigned short* sw1 = (unsigned short*)(ws + OFF_W1);
  unsigned short* sw2 = (unsigned short*)(ws + OFF_W2);
  unsigned short* swe = (unsigned short*)(ws + OFF_WE);
  float* sc = (float*)(ws + OFF_SC);
  float* bi = (float*)(ws + OFF_BI);
  float* bc = (float*)(ws + OFF_B2);

  if (idx < 8192) {                                   // W1: [H=128, D=64]
    int i = idx & 15, nl = (idx >> 4) & 15, kh = (idx >> 8) & 1, ks = (idx >> 9) & 1, nt = idx >> 10;
    int K = ks * 32 + ((i >> 3) << 4) + kh * 8 + (i & 7);
    int col = nt * 16 + nl;
    sw1[idx] = f2bf(W1[col * D_ + K]);
  } else if (idx < 8192 + 2048) {                     // W2: [E=8, H=128], pad cols to 16
    int j = idx - 8192;
    int i = j & 15, nl = (j >> 4) & 15, kh = (j >> 8) & 1, ks = j >> 9;
    int K = ks * 32 + ((i >> 3) << 4) + kh * 8 + (i & 7);
    sw2[j] = (nl < E_) ? f2bf(W2[nl * H_ + K]) : (unsigned short)0;
  } else if (idx < 8192 + 2048 + 9216) {              // We: [144, 64] flat
    int j = idx - (8192 + 2048);
    int i = j & 15, nl = (j >> 4) & 15, kh = (j >> 8) & 1, ks = (j >> 9) & 1, nt = j >> 10;
    int K = ks * 32 + ((i >> 3) << 4) + kh * 8 + (i & 7);
    int jj = nt * 16 + nl;                            // always < 144
    swe[j] = f2bf(We[jj * D_ + K]);
  } else if (idx < 8192 + 2048 + 9216 + 128) {        // BN fold (eval mode, running stats)
    int h = idx - (8192 + 2048 + 9216);
    float s = gamma[h] * rsqrtf(rvar[h] + 1e-5f);
    sc[h] = s;
    bi[h] = beta[h] + (b1[h] - rmean[h]) * s;
  } else if (idx < 8192 + 2048 + 9216 + 128 + 16) {   // b2 padded
    int e = idx - (8192 + 2048 + 9216 + 128);
    bc[e] = (e < E_) ? b2[e] : 0.f;
  }
}

// ---------------- fused MoE kernel ----------------
__global__ __launch_bounds__(NTHREADS)
void moe_fused(const float* __restrict__ X, const unsigned char* __restrict__ wsb,
               float* __restrict__ outA, float* __restrict__ outRet,
               int tiles, int tpw) {
  __shared__ __align__(16) unsigned char sW[WBYTES];
  __shared__ __align__(16) float sEO[NW][16 * JTOT];   // per-wave: h-stage (bf16) then weighted expert outs
  __shared__ __align__(16) float sRet[NW][16 * E_];    // per-wave routing weights

  // ---- stage swizzled weights into LDS once per block (async, no VGPR round-trip) ----
  {
    const unsigned swb = (unsigned)(unsigned long long)(void*)sW;  // LDS byte offset (low 32 bits of generic ptr)
    for (int i = threadIdx.x; i < WBYTES / 16; i += NTHREADS)
      async_copy16(swb + i * 16u, wsb + i * 16);
    wait_async0();
  }
  __syncthreads();

  const unsigned short* sW1 = (const unsigned short*)(sW + OFF_W1);
  const unsigned short* sW2 = (const unsigned short*)(sW + OFF_W2);
  const unsigned short* sWe = (const unsigned short*)(sW + OFF_WE);
  const float* sSc = (const float*)(sW + OFF_SC);
  const float* sBi = (const float*)(sW + OFF_BI);
  const float* sB2 = (const float*)(sW + OFF_B2);

  const int lane = threadIdx.x & 31;
  const int wv   = threadIdx.x >> 5;
  const int ln   = lane & 15;   // column / A-row index
  const int kh   = lane >> 4;   // K-half selector

  unsigned short* sH  = (unsigned short*)&sEO[wv][0];  // 16x128 bf16 (aliases sEO, used first)
  float* eoBuf  = &sEO[wv][0];
  float* retBuf = &sRet[wv][0];

  const int t0 = (blockIdx.x * NW + wv) * tpw;
  for (int t = t0; t < t0 + tpw; ++t) {
    if (t >= tiles) break;
    const int row0 = t * 16;
    const float* xt = X + (size_t)row0 * D_;

    if (t + 1 < tiles)  // pull next X tile toward the caches
      __builtin_prefetch(X + (size_t)(row0 + 16) * D_ + lane * 32, 0, 0);

    // ---- A fragments of X tile (M = ln, K-half = kh) ----
    const float* xr = xt + (size_t)ln * D_;
    bf16x16 a0 = make_afrag(xr, 0, kh);
    bf16x16 a1 = make_afrag(xr, 32, kh);

    // ---- GEMM1: h = X @ W1^T, + BN + ReLU, stage bf16 to LDS ----
#pragma unroll
    for (int nt = 0; nt < 8; ++nt) {
      f32x8 acc = {};
      bf16x16 b0 = load_frag16(sW1 + boff(nt * 2 + 0, kh, ln));
      acc = __builtin_amdgcn_wmma_f32_16x16x32_bf16(false, a0, false, b0, (short)0, acc, false, false);
      bf16x16 b1f = load_frag16(sW1 + boff(nt * 2 + 1, kh, ln));
      acc = __builtin_amdgcn_wmma_f32_16x16x32_bf16(false, a1, false, b1f, (short)0, acc, false, false);
      const int col = nt * 16 + ln;
      const float s = sSc[col], bb = sBi[col];
#pragma unroll
      for (int r = 0; r < 8; ++r) {
        const int m = r + 8 * kh;
        float h = fmaxf(fmaf(acc[r], s, bb), 0.f);
        sH[m * H_ + col] = f2bf(h);
      }
    }

    // ---- GEMM2: logits = h @ W2^T (cols 8..15 padded zero) ----
    f32x8 lacc = {};
#pragma unroll
    for (int ks = 0; ks < 4; ++ks) {
      bf16x16 ah = load_afrag_lds(sH, ln, ks, kh);
      bf16x16 bw = load_frag16(sW2 + boff(ks, kh, ln));
      lacc = __builtin_amdgcn_wmma_f32_16x16x32_bf16(false, ah, false, bw, (short)0, lacc, false, false);
    }

    // ---- softmax + straight-through hard routing (8 experts per row) ----
    const float b2v = sB2[ln];
#pragma unroll
    for (int r = 0; r < 8; ++r) {
      const int m = r + 8 * kh;
      float v = lacc[r] + b2v;
      float mx = v;
      mx = fmaxf(mx, __shfl_xor(mx, 1, 32));
      mx = fmaxf(mx, __shfl_xor(mx, 2, 32));
      mx = fmaxf(mx, __shfl_xor(mx, 4, 32));
      float ex = __expf(v - mx);
      float sm = ex;
      sm += __shfl_xor(sm, 1, 32);
      sm += __shfl_xor(sm, 2, 32);
      sm += __shfl_xor(sm, 4, 32);
      float ys = ex / sm;
      int cand = (v == mx) ? (ln & 7) : 8;              // first-max tie break, like jnp.argmax
      int c1 = __shfl_xor(cand, 1, 32); cand = cand < c1 ? cand : c1;
      int c2 = __shfl_xor(cand, 2, 32); cand = cand < c2 ? cand : c2;
      int c4 = __shfl_xor(cand, 4, 32); cand = cand < c4 ? cand : c4;
      float yh = ((ln & 7) == cand) ? 1.f : 0.f;
      float rv = (yh - ys) + ys;                        // exact 0 off-argmax, ~1 at argmax
      if (ln < 8) {
        retBuf[m * E_ + ln] = rv;
        outRet[(size_t)(row0 + m) * E_ + ln] = rv;
      }
    }

    // ---- all-expert GEMM: eo = X @ We^T (144 cols), weight by ret into LDS ----
#pragma unroll
    for (int j = 0; j < NTJ; ++j) {
      f32x8 ea = {};
      bf16x16 be0 = load_frag16(sWe + boff(j * 2 + 0, kh, ln));
      ea = __builtin_amdgcn_wmma_f32_16x16x32_bf16(false, a0, false, be0, (short)0, ea, false, false);
      bf16x16 be1 = load_frag16(sWe + boff(j * 2 + 1, kh, ln));
      ea = __builtin_amdgcn_wmma_f32_16x16x32_bf16(false, a1, false, be1, (short)0, ea, false, false);
      const int jj = j * 16 + ln;                       // 0..143
      const int e = jj / A_;
#pragma unroll
      for (int r = 0; r < 8; ++r) {
        const int m = r + 8 * kh;
        eoBuf[m * JTOT + jj] = retBuf[m * E_ + e] * ea[r];
      }
    }

    // ---- gather: out[m,a] = sum_e eo_w[m, e*18+a] ----
#pragma unroll
    for (int p = 0; p < 9; ++p) {
      const int q = p * 32 + lane;                      // 0..287 = 16 rows * 18 outs
      const int m = q / A_;
      const int a = q - m * A_;
      float s = 0.f;
#pragma unroll
      for (int e = 0; e < E_; ++e) s += eoBuf[m * JTOT + e * A_ + a];
      outA[(size_t)(row0 + m) * A_ + a] = s;
    }
  }
}

// ---------------- launcher ----------------
extern "C" void kernel_launch(void* const* d_in, const int* in_sizes, int n_in,
                              void* d_out, int out_size, void* d_ws, size_t ws_size,
                              hipStream_t stream) {
  const float* X     = (const float*)d_in[0];
  const float* W1    = (const float*)d_in[1];
  const float* b1    = (const float*)d_in[2];
  const float* gamma = (const float*)d_in[3];
  const float* beta  = (const float*)d_in[4];
  const float* rmean = (const float*)d_in[5];
  const float* rvar  = (const float*)d_in[6];
  const float* W2    = (const float*)d_in[7];
  const float* b2    = (const float*)d_in[8];
  const float* We    = (const float*)d_in[9];

  const int N = in_sizes[0] / D_;               // 524288
  float* out    = (float*)d_out;
  float* outA   = out;                          // [N, 18]
  float* outRet = out + (size_t)N * A_;         // [N, 8]
  unsigned char* ws = (unsigned char*)d_ws;

  const int prep_items = 8192 + 2048 + 9216 + 128 + 16;
  moe_prep<<<(prep_items + 255) / 256, 256, 0, stream>>>(W1, b1, gamma, beta, rmean, rvar, W2, b2, We, ws);

  const int tiles = N / 16;                     // 32768 (N is a multiple of 16)
  const int blocks = 2048;
  const int tpw = (tiles + blocks * NW - 1) / (blocks * NW);
  moe_fused<<<blocks, NTHREADS, 0, stream>>>(X, ws, outA, outRet, tiles, tpw);
}
